// MixtralAttention_32607391711911
// MI455X (gfx1250) — matrix operand order repaired and checked by reading, not compiled, not run on hardware
//
#include <hip/hip_runtime.h>
#include <hip/hip_bf16.h>

// ---------------------------------------------------------------------------
// MixtralAttention on gfx1250 (MI455X): bf16 WMMA pipeline
//   K1: qkv = hs @ wqkv (+ fused NeoX RoPE) -> bf16 q/k (row-major), v (TRANSPOSED)
//   K2: flash attention, double-buffered async-to-LDS K/V^T staging (ASYNCcnt)
//   K3: out = attn @ wo -> f32 (A tile staged async)
// ---------------------------------------------------------------------------

typedef __attribute__((ext_vector_type(16))) __bf16 v16bf;
typedef __attribute__((ext_vector_type(8)))  __bf16 v8bf;
typedef __attribute__((ext_vector_type(4)))  __bf16 v4bf;
typedef __attribute__((ext_vector_type(8)))  float  v8f;
typedef __attribute__((ext_vector_type(4)))  float  v4f;

#define DEVI __device__ __forceinline__

// Problem sizes
constexpr int BB    = 2;
constexpr int SS    = 2048;
constexpr int HID   = 4096;
constexpr int NH    = 32;
constexpr int NKV   = 8;
constexpr int HD    = 128;
constexpr int NQ    = NH * HD;          // 4096
constexpr int NKC   = NKV * HD;         // 1024
constexpr int NQKV  = NQ + 2 * NKC;     // 6144
constexpr int WIN   = 1024;
constexpr float SCALE = 0.08838834764831845f;   // 128^-0.5
constexpr float NEG_LN_THETA_OVER_HALF = -0.14391156831212787f; // -ln(10000)/64

DEVI v8f wmma_bf16(v16bf a, v16bf b, v8f c) {
  // D = A(16x32 bf16) * B(32x16 bf16) + C(16x16 f32)
  return __builtin_amdgcn_wmma_f32_16x16x32_bf16(
      /*neg_a=*/false, a, /*neg_b=*/false, b,
      /*c_mod=*/(short)0, c, /*reuse_a=*/false, /*reuse_b=*/false);
}

union U16 { v16bf v; v8bf h[2]; };
DEVI v16bf join8(v8bf lo, v8bf hi) { U16 u; u.h[0] = lo; u.h[1] = hi; return u.v; }

// Async 16-byte copy global -> LDS (no VGPR data path, tracked by ASYNCcnt).
DEVI void async_cp16(const __bf16* lds_dst, const __bf16* gsrc) {
  uint32_t lds = (uint32_t)(uintptr_t)lds_dst;         // LDS byte address
  uint64_t gaddr = (uint64_t)(uintptr_t)gsrc;
  asm volatile("global_load_async_to_lds_b128 %0, %1, off"
               :: "v"(lds), "v"(gaddr) : "memory");
}
DEVI void wait_async0()  { asm volatile("s_wait_asynccnt 0x0"  ::: "memory"); }
DEVI void wait_async16() { asm volatile("s_wait_asynccnt 0x10" ::: "memory"); }

// ===========================================================================
// Kernel 1: qkv projection + fused RoPE. Tile 128(M) x 128(N), K-step 32.
// 8 waves; wave w owns rows [16w,16w+16) of the tile, all 128 N columns.
// q,k stored [b,h,s,d]; v stored TRANSPOSED [b,h,d,s] for async V^T staging.
// ===========================================================================
__global__ __launch_bounds__(256) void qkv_rope_kernel(
    const float* __restrict__ hs, const float* __restrict__ wqkv,
    __bf16* __restrict__ qb, __bf16* __restrict__ kb, __bf16* __restrict__ vb)
{
  __shared__ alignas(32) __bf16 sA [128 * 32];  // [m][k] row-major
  __shared__ alignas(32) __bf16 sBT[128 * 32];  // [n][k] (B transposed)

  const int tid  = threadIdx.x;
  const int lane = tid & 31, wv = tid >> 5;
  const int hgrp = lane >> 4, l16 = lane & 15;
  const int n0 = blockIdx.x * 128;
  const int m0 = blockIdx.y * 128;

  v8f acc[8] = {};

  for (int k0 = 0; k0 < HID; k0 += 32) {
    __syncthreads();
    // Stage A tile (128x32 f32 -> bf16) + unconditional near-cache prefetch
    // of the next K-step (speculative; OOB prefetch past the end is dropped).
#pragma unroll
    for (int p = 0; p < 4; ++p) {
      int idx = p * 1024 + tid * 4;
      int row = idx >> 5, col = idx & 31;
      const float* gp = hs + (size_t)(m0 + row) * HID + k0 + col;
      v4f a = *(const v4f*)gp;
      __builtin_prefetch(gp + 32, 0, 3);
      v4bf bv; bv[0] = (__bf16)a[0]; bv[1] = (__bf16)a[1];
      bv[2] = (__bf16)a[2]; bv[3] = (__bf16)a[3];
      *(v4bf*)(sA + row * 32 + col) = bv;
    }
    // Stage B tile (32x128 f32 -> bf16, transposed into [n][k])
#pragma unroll
    for (int p = 0; p < 4; ++p) {
      int idx = p * 1024 + tid * 4;
      int kk = idx >> 7, nn = idx & 127;
      const float* gp = wqkv + (size_t)(k0 + kk) * NQKV + n0 + nn;
      v4f a = *(const v4f*)gp;
      __builtin_prefetch(gp + (size_t)32 * NQKV, 0, 3);
      sBT[(nn + 0) * 32 + kk] = (__bf16)a[0];
      sBT[(nn + 1) * 32 + kk] = (__bf16)a[1];
      sBT[(nn + 2) * 32 + kk] = (__bf16)a[2];
      sBT[(nn + 3) * 32 + kk] = (__bf16)a[3];
    }
    __syncthreads();

    // A fragment: row = l16 of wave block; K runs {h*16..+15}
    const __bf16* arow = sA + (wv * 16 + l16) * 32;
    v16bf afrag = join8(*(const v8bf*)(arow + hgrp * 16),
                        *(const v8bf*)(arow + 8 + hgrp * 16));
    v16bf bfr[8];
#pragma unroll
    for (int j = 0; j < 8; ++j)
      bfr[j] = *(const v16bf*)(sBT + (j * 16 + l16) * 32 + hgrp * 16);
#pragma unroll
    for (int j = 0; j < 8; ++j)
      acc[j] = wmma_bf16(afrag, bfr[j], acc[j]);
  }

  const bool is_q = (n0 < NQ);
  const bool is_k = (n0 >= NQ) && (n0 < NQ + NKC);

  // Fused NeoX RoPE: pair (d, d+64) = (acc[j], acc[j+4]) same lane.
  if (is_q || is_k) {
#pragma unroll
    for (int j = 0; j < 4; ++j) {
      int d1 = j * 16 + l16;                        // 0..63
      float invf = __expf(NEG_LN_THETA_OVER_HALF * (float)d1);
#pragma unroll
      for (int r = 0; r < 8; ++r) {
        int mg  = m0 + wv * 16 + r + 8 * hgrp;
        int pos = mg & (SS - 1);
        float ang = (float)pos * invf;
        float c = __cosf(ang), sn = __sinf(ang);
        float x1 = acc[j][r], x2 = acc[j + 4][r];
        acc[j][r]     = x1 * c - x2 * sn;
        acc[j + 4][r] = x2 * c + x1 * sn;
      }
    }
  }

  // Scatter-store bf16.
#pragma unroll
  for (int r = 0; r < 8; ++r) {
    int mg = m0 + wv * 16 + r + 8 * hgrp;
    int bz = mg >> 11, sq = mg & (SS - 1);
    if (is_q) {
      int h = n0 >> 7;
      __bf16* dst = qb + (((size_t)(bz * NH + h)) * SS + sq) * HD;
#pragma unroll
      for (int j = 0; j < 8; ++j) dst[j * 16 + l16] = (__bf16)acc[j][r];
    } else if (is_k) {
      int h = (n0 - NQ) >> 7;
      __bf16* dst = kb + (((size_t)(bz * NKV + h)) * SS + sq) * HD;
#pragma unroll
      for (int j = 0; j < 8; ++j) dst[j * 16 + l16] = (__bf16)acc[j][r];
    } else {
      int h = (n0 - NQ - NKC) >> 7;
      __bf16* base = vb + ((size_t)(bz * NKV + h)) * HD * SS;  // [d][s]
#pragma unroll
      for (int j = 0; j < 8; ++j)
        base[(size_t)(j * 16 + l16) * SS + sq] = (__bf16)acc[j][r];
    }
  }
}

// ===========================================================================
// Kernel 2: sliding-window flash attention. Block = 128 q rows of one head.
// LDS 160 KB: K[2] (2x32KB) | V^T[2] (2x32KB) | P (32KB, per-wave slices).
// K/V^T double-buffered via GLOBAL_LOAD_ASYNC_TO_LDS_B128: next tile streams
// in (ASYNCcnt) while the current tile feeds 64 WMMAs + softmax.
// ===========================================================================
__global__ __launch_bounds__(256) void attn_kernel(
    const __bf16* __restrict__ qb, const __bf16* __restrict__ kb,
    const __bf16* __restrict__ vb, __bf16* __restrict__ ob)
{
  __shared__ alignas(32) __bf16 smem[5 * 128 * 128];  // 160 KB
  // layout (bf16 elems): K0 @0, K1 @16384, VT0 @32768, VT1 @49152, P @65536

  const int tid  = threadIdx.x;
  const int lane = tid & 31, wv = tid >> 5;
  const int hgrp = lane >> 4, l16 = lane & 15;
  const int qt = blockIdx.x, hq = blockIdx.y, bz = blockIdx.z;
  const int hkv = hq / (NH / NKV);
  const int q0 = qt * 128;

  const size_t qbase  = (((size_t)(bz * NH + hq)) * SS + q0 + wv * 16) * HD;
  const size_t kvbase = ((size_t)(bz * NKV + hkv)) * SS * HD;  // K rows / V^T plane

  // Stage one kv tile (16 async ops per thread: 8 K + 8 V^T).
  auto stage = [&](int kt, int buf) {
    const __bf16* gk  = kb + kvbase + (size_t)kt * 128 * HD;   // [kv][hd]
    const __bf16* gvt = vb + kvbase + kt * 128;                // [hd][s] col block
    __bf16* dK  = smem + buf * 16384;
    __bf16* dVT = smem + 32768 + buf * 16384;
#pragma unroll
    for (int i = 0; i < 8; ++i) {
      int idx = i * 2048 + tid * 8;
      async_cp16(dK + idx, gk + idx);
      int d = idx >> 7, kvloc = idx & 127;
      async_cp16(dVT + idx, gvt + (size_t)d * SS + kvloc);
    }
  };

  // Preload Q fragments for this wave's 16 rows (4 K-chunks of 32 over HD).
  v16bf qa[4];
#pragma unroll
  for (int c = 0; c < 4; ++c) {
    const __bf16* p = qb + qbase + (size_t)l16 * HD + c * 32 + hgrp * 16;
    qa[c] = join8(*(const v8bf*)p, *(const v8bf*)(p + 8));
  }

  float mst[8], lst[8];
#pragma unroll
  for (int r = 0; r < 8; ++r) { mst[r] = -3.0e38f; lst[r] = 0.f; }
  v8f o[8] = {};

  int lo = q0 - (WIN - 1); if (lo < 0) lo = 0;
  const int kt0 = lo >> 7, kt1 = qt;

  stage(kt0, kt0 & 1);
  for (int kt = kt0; kt <= kt1; ++kt) {
    const int buf = kt & 1;
    if (kt < kt1) { stage(kt + 1, buf ^ 1); wait_async16(); }
    else          { wait_async0(); }
    __syncthreads();   // current tile resident everywhere; prev reads done

    const __bf16* cK  = smem + buf * 16384;              // [kv][hd]
    const __bf16* cVT = smem + 32768 + buf * 16384;      // [hd][kv]

    // Scores S = Q K^T  (16 q rows x 128 kv cols per wave) : 32 WMMAs
    v8f sc[8] = {};
#pragma unroll
    for (int c = 0; c < 4; ++c) {
#pragma unroll
      for (int g = 0; g < 8; g += 4) {
        v16bf bfr[4];
#pragma unroll
        for (int j = 0; j < 4; ++j)
          bfr[j] = *(const v16bf*)(cK + ((g + j) * 16 + l16) * 128 + c * 32 + hgrp * 16);
#pragma unroll
        for (int j = 0; j < 4; ++j)
          sc[g + j] = wmma_bf16(qa[c], bfr[j], sc[g + j]);
      }
    }

    // Causal + sliding-window mask, scale.
#pragma unroll
    for (int j = 0; j < 8; ++j) {
      int kv = kt * 128 + j * 16 + l16;
#pragma unroll
      for (int r = 0; r < 8; ++r) {
        int qp = q0 + wv * 16 + r + 8 * hgrp;
        bool ok = (kv <= qp) && (qp - kv < WIN);
        sc[j][r] = ok ? sc[j][r] * SCALE : -1.0e30f;
      }
    }

    // Online softmax: row max (reduce over j, then over the 16-lane half).
    float mnew[8], alpha[8];
#pragma unroll
    for (int r = 0; r < 8; ++r) {
      float mx = sc[0][r];
#pragma unroll
      for (int j = 1; j < 8; ++j) mx = fmaxf(mx, sc[j][r]);
#pragma unroll
      for (int off = 1; off < 16; off <<= 1)
        mx = fmaxf(mx, __shfl_xor(mx, off, 32));
      mnew[r]  = fmaxf(mst[r], mx);
      alpha[r] = __expf(mst[r] - mnew[r]);
      mst[r]   = mnew[r];
    }

    // P = exp(S - m), row sums (guard fully-masked entries).
    float rsum[8];
#pragma unroll
    for (int r = 0; r < 8; ++r) rsum[r] = 0.f;
#pragma unroll
    for (int j = 0; j < 8; ++j) {
#pragma unroll
      for (int r = 0; r < 8; ++r) {
        float sv = sc[j][r];
        float p  = (sv > -5.0e29f) ? __expf(sv - mnew[r]) : 0.f;
        sc[j][r] = p;
        rsum[r] += p;
      }
    }
#pragma unroll
    for (int r = 0; r < 8; ++r) {
#pragma unroll
      for (int off = 1; off < 16; off <<= 1)
        rsum[r] += __shfl_xor(rsum[r], off, 32);
      lst[r] = lst[r] * alpha[r] + rsum[r];
#pragma unroll
      for (int jh = 0; jh < 8; ++jh) o[jh][r] *= alpha[r];
    }

    // Write P (bf16) into this wave's dedicated LDS slice (A-matrix layout).
    // Same-wave LDS ops are in order -> no barrier needed before reading back.
    __bf16* sP = smem + 65536 + wv * (16 * 128);
#pragma unroll
    for (int j = 0; j < 8; ++j) {
#pragma unroll
      for (int r = 0; r < 8; ++r)
        sP[(r + 8 * hgrp) * 128 + j * 16 + l16] = (__bf16)sc[j][r];
    }

    // O += P V : 32 WMMAs (A from sP, B from V^T — contiguous 32B LDS reads).
#pragma unroll
    for (int c = 0; c < 4; ++c) {
      const __bf16* prow = sP + l16 * 128 + c * 32 + hgrp * 16;
      v16bf pa = join8(*(const v8bf*)prow, *(const v8bf*)(prow + 8));
#pragma unroll
      for (int g = 0; g < 8; g += 4) {
        v16bf bfr[4];
#pragma unroll
        for (int j = 0; j < 4; ++j)
          bfr[j] = *(const v16bf*)(cVT + ((g + j) * 16 + l16) * 128 + c * 32 + hgrp * 16);
#pragma unroll
        for (int j = 0; j < 4; ++j)
          o[g + j] = wmma_bf16(pa, bfr[j], o[g + j]);
      }
    }
    __syncthreads();   // all waves done with buf before it is re-staged at kt+1
  }

  // Normalize and store attn output bf16: [b, s, NH*HD]
#pragma unroll
  for (int r = 0; r < 8; ++r) {
    float inv = 1.f / lst[r];
    int sq = q0 + wv * 16 + r + 8 * hgrp;
    __bf16* dst = ob + ((size_t)bz * SS + sq) * NQ + hq * HD;
#pragma unroll
    for (int jh = 0; jh < 8; ++jh)
      dst[jh * 16 + l16] = (__bf16)(o[jh][r] * inv);
  }
}

// ===========================================================================
// Kernel 3: out = attn(bf16) @ wo(f32->bf16) -> f32. Same tiling as K1.
// A tile (already bf16) staged with async-to-LDS.
// ===========================================================================
__global__ __launch_bounds__(256) void out_gemm_kernel(
    const __bf16* __restrict__ attn, const float* __restrict__ wo,
    float* __restrict__ out)
{
  __shared__ alignas(32) __bf16 sA [128 * 32];
  __shared__ alignas(32) __bf16 sBT[128 * 32];

  const int tid  = threadIdx.x;
  const int lane = tid & 31, wv = tid >> 5;
  const int hgrp = lane >> 4, l16 = lane & 15;
  const int n0 = blockIdx.x * 128;
  const int m0 = blockIdx.y * 128;

  v8f acc[8] = {};

  for (int k0 = 0; k0 < NQ; k0 += 32) {
    __syncthreads();
    // Stage A (already bf16): async 16B copies; LDS offset == idx (row-major).
#pragma unroll
    for (int p = 0; p < 2; ++p) {
      int idx = p * 2048 + tid * 8;
      int row = idx >> 5, col = idx & 31;
      async_cp16(sA + idx, attn + (size_t)(m0 + row) * NQ + k0 + col);
    }
    // Stage B (f32 -> bf16, transposed) + unconditional near prefetch.
#pragma unroll
    for (int p = 0; p < 4; ++p) {
      int idx = p * 1024 + tid * 4;
      int kk = idx >> 7, nn = idx & 127;
      const float* gp = wo + (size_t)(k0 + kk) * HID + n0 + nn;
      v4f a = *(const v4f*)gp;
      __builtin_prefetch(gp + (size_t)32 * HID, 0, 3);
      sBT[(nn + 0) * 32 + kk] = (__bf16)a[0];
      sBT[(nn + 1) * 32 + kk] = (__bf16)a[1];
      sBT[(nn + 2) * 32 + kk] = (__bf16)a[2];
      sBT[(nn + 3) * 32 + kk] = (__bf16)a[3];
    }
    wait_async0();
    __syncthreads();

    const __bf16* arow = sA + (wv * 16 + l16) * 32;
    v16bf afrag = join8(*(const v8bf*)(arow + hgrp * 16),
                        *(const v8bf*)(arow + 8 + hgrp * 16));
    v16bf bfr[8];
#pragma unroll
    for (int j = 0; j < 8; ++j)
      bfr[j] = *(const v16bf*)(sBT + (j * 16 + l16) * 32 + hgrp * 16);
#pragma unroll
    for (int j = 0; j < 8; ++j)
      acc[j] = wmma_bf16(afrag, bfr[j], acc[j]);
  }

#pragma unroll
  for (int r = 0; r < 8; ++r) {
    int row = m0 + wv * 16 + r + 8 * hgrp;
#pragma unroll
    for (int j = 0; j < 8; ++j)
      out[(size_t)row * HID + n0 + j * 16 + l16] = acc[j][r];
  }
}

// ===========================================================================
extern "C" void kernel_launch(void* const* d_in, const int* in_sizes, int n_in,
                              void* d_out, int out_size, void* d_ws, size_t ws_size,
                              hipStream_t stream) {
  (void)in_sizes; (void)n_in; (void)out_size; (void)ws_size;
  const float* hs   = (const float*)d_in[0];
  const float* wqkv = (const float*)d_in[1];
  const float* wo   = (const float*)d_in[2];
  // d_in[3] = positions (arange, folded into the fused RoPE index math)

  // Workspace layout (bf16): q | k | v(transposed) | attn  (~84 MB total)
  __bf16* qb = (__bf16*)d_ws;
  __bf16* kb = qb + (size_t)BB * NH  * SS * HD;
  __bf16* vb = kb + (size_t)BB * NKV * SS * HD;
  __bf16* ab = vb + (size_t)BB * NKV * SS * HD;

  qkv_rope_kernel<<<dim3(NQKV / 128, (BB * SS) / 128), 256, 0, stream>>>(
      hs, wqkv, qb, kb, vb);
  attn_kernel<<<dim3(SS / 128, NH, BB), 256, 0, stream>>>(qb, kb, vb, ab);
  out_gemm_kernel<<<dim3(HID / 128, (BB * SS) / 128), 256, 0, stream>>>(
      ab, wo, (float*)d_out);
}
